// MultiHeadAttention_39573828665699
// MI455X (gfx1250) — matrix operand order repaired and checked by reading, not compiled, not run on hardware
//
#include <hip/hip_runtime.h>
#include <hip/hip_bf16.h>
#include <math.h>

// ---------------------------------------------------------------------------
// MultiHeadAttention forward for MI455X (gfx1250, wave32, WMMA bf16 path)
//   B=8, S=1024, D_MODEL=512, H=8, DEPTH=64
// d_out = out [B,S,512] f32  ||  attn [B,H,S,S] f32
// Roofline: ~35 GFLOP, ~600 MB of mandatory attn/weights traffic (~26 us at
// 23.3 TB/s). bf16 WMMA (f32 accum) pushes compute well under the memory
// time; the kernels below are organized to keep the WMMA pipes fed from LDS.
// ---------------------------------------------------------------------------

#define Bn 8
#define Sn 1024
#define Dm 512
#define Hn 8
#define Dp 64

typedef __bf16 bf16_t;
typedef bf16_t v16bf  __attribute__((ext_vector_type(16)));
typedef bf16_t bf16x8 __attribute__((ext_vector_type(8)));
typedef bf16_t bf16x4 __attribute__((ext_vector_type(4)));
typedef float  v8f    __attribute__((ext_vector_type(8)));
typedef float  f32x4  __attribute__((ext_vector_type(4)));
typedef unsigned int u32x4 __attribute__((ext_vector_type(4)));
typedef int    i32x8  __attribute__((ext_vector_type(8)));
typedef int    i32x4  __attribute__((ext_vector_type(4)));

#if defined(__has_builtin)
#if __has_builtin(__builtin_amdgcn_tensor_load_to_lds) && \
    __has_builtin(__builtin_amdgcn_s_wait_tensorcnt)
#define USE_TDM 1
#endif
#endif

__device__ __forceinline__ v8f wmma_bf16(v16bf a, v16bf b, v8f c) {
  return __builtin_amdgcn_wmma_f32_16x16x32_bf16(
      false, a, false, b, (short)0, c, false, false);
}

__device__ __forceinline__ v8f zero_v8f() {
  v8f z;
#pragma unroll
  for (int i = 0; i < 8; ++i) z[i] = 0.0f;
  return z;
}

// Build a 16-element bf16 fragment from two 16B-aligned 8-element chunks.
__device__ __forceinline__ v16bf frag_from_2x8(const bf16_t* p0,
                                               const bf16_t* p1) {
  bf16x8 a = *(const bf16x8*)p0;
  bf16x8 b = *(const bf16x8*)p1;
  return __builtin_shufflevector(a, b, 0, 1, 2, 3, 4, 5, 6, 7,
                                 8, 9, 10, 11, 12, 13, 14, 15);
}

// ---------------------------------------------------------------------------
// GEMM + bias:  C[M,N] = A[M,K] * W[K,N] + bias[N]   (A,W,bias f32)
// Block = 256 thr / 8 waves computes a 128x64 tile; k stepped by 32.
// A and W(transposed) staged in LDS as bf16; fragments via 16B LDS loads.
//   headSplit==0 : f32 row-major to outF
//   headSplit==1 : bf16 head-split [B,H,S,Dp] to outBF
// ---------------------------------------------------------------------------
#define GM 128
#define GN 64
#define LPAD 40  // LDS row stride (bf16 elems) for 32-wide tiles: conflict-free

__global__ __launch_bounds__(256) void mha_gemm_bias_kernel(
    const float* __restrict__ A, const float* __restrict__ W,
    const float* __restrict__ bias, float* __restrict__ outF,
    bf16_t* __restrict__ outBF, int M, int Kd, int N, int headSplit) {
  __shared__ bf16_t Asub[GM * LPAD];  // 128 rows x 32 k (stride 40)
  __shared__ bf16_t Wsub[GN * LPAD];  // 64 cols  x 32 k (transposed, stride 40)

  const int tid = threadIdx.x;
  const int wave = tid >> 5;
  const int lane = tid & 31;
  const int lr = lane & 15, lh = lane >> 4;
  const int tm = blockIdx.x * GM;
  const int tn = blockIdx.y * GN;

  v8f acc[4];
#pragma unroll
  for (int n = 0; n < 4; ++n) acc[n] = zero_v8f();

  for (int k0 = 0; k0 < Kd; k0 += 32) {
    // ---- cooperative stage: A block 128x32 (f32 -> bf16) ----
#pragma unroll
    for (int j = 0; j < 4; ++j) {
      const int c = tid + 256 * j;  // 1024 float4 chunks
      const int row = c >> 3, seg = c & 7;
      f32x4 v = *(const f32x4*)(A + (size_t)(tm + row) * Kd + k0 + seg * 4);
      bf16x4 h;
      h[0] = (bf16_t)v[0]; h[1] = (bf16_t)v[1];
      h[2] = (bf16_t)v[2]; h[3] = (bf16_t)v[3];
      *(bf16x4*)(Asub + row * LPAD + seg * 4) = h;
    }
    // ---- cooperative stage: W block 32x64, stored transposed [n][k] ----
#pragma unroll
    for (int j = 0; j < 2; ++j) {
      const int c = tid + 256 * j;  // 512 float4 chunks
      const int kk = c >> 4, seg = c & 15;
      f32x4 v = *(const f32x4*)(W + (size_t)(k0 + kk) * N + tn + seg * 4);
#pragma unroll
      for (int e = 0; e < 4; ++e)
        Wsub[(seg * 4 + e) * LPAD + kk] = (bf16_t)v[e];
    }
    // prefetch next A k-slice (global_prefetch_b8)
    if (k0 + 32 < Kd)
      __builtin_prefetch(
          A + (size_t)(tm + (tid >> 1)) * Kd + (k0 + 32) + (tid & 1) * 16, 0, 1);
    __syncthreads();

    // A fragment: lane row = 16*wave + lr, K chunk = lh*16..+15 (contiguous)
    const bf16_t* arow = Asub + (wave * 16 + lr) * LPAD + lh * 16;
    v16bf a = frag_from_2x8(arow, arow + 8);
#pragma unroll
    for (int nt = 0; nt < 4; ++nt) {
      // B fragment: lane col = nt*16 + lr, K chunk = lh*16..+15 (contiguous)
      const bf16_t* bp = Wsub + (nt * 16 + lr) * LPAD + lh * 16;
      v16bf b = frag_from_2x8(bp, bp + 8);
      acc[nt] = wmma_bf16(a, b, acc[nt]);
    }
    __syncthreads();
  }

  const int row0 = tm + wave * 16 + lh * 8;
#pragma unroll
  for (int nt = 0; nt < 4; ++nt) {
    const int col = tn + nt * 16 + lr;
    const float bv = bias[col];
#pragma unroll
    for (int r = 0; r < 8; ++r) {
      const int row = row0 + r;
      const float val = acc[nt][r] + bv;
      if (headSplit) {
        const int bb = row >> 10, ss = row & (Sn - 1);
        const int hh = col >> 6, dd = col & (Dp - 1);
        outBF[(((size_t)bb * Hn + hh) * Sn + ss) * Dp + dd] = (bf16_t)val;
      } else {
        outF[(size_t)row * N + col] = val;
      }
    }
  }
}

// ---------------------------------------------------------------------------
// Fused attention: one block = (b, h, 16-query-row tile), 256 thr / 8 waves.
// LDS map (dynamic):
//   [0,64K)      sc    f32 16x1024 score strip
//   [64K,96K)    abf   bf16 16x1024 attn copy for the AV GEMM
//   [96K,104K)   part  f32 8x16x16 wave partials
//   [104K,105K)  red   f32 256 softmax scratch
//   [105K,233K)  vst   bf16 1024x64 V head, filled by the Tensor Data Mover
// Only possible on CDNA5's 320KB-per-workgroup LDS.
// ---------------------------------------------------------------------------
#define SC_OFF 0
#define ABF_OFF 65536
#define PART_OFF (ABF_OFF + 32768)
#define RED_OFF (PART_OFF + 8192)
#define VST_OFF (RED_OFF + 1024)
#define SMEM_TOTAL (VST_OFF + Sn * Dp * 2)  // 238592 bytes

__global__ __launch_bounds__(256) void mha_attention_kernel(
    const bf16_t* __restrict__ Q, const bf16_t* __restrict__ Kmat,
    const bf16_t* __restrict__ V, const float* __restrict__ mask,
    const float* __restrict__ addw, float* __restrict__ attn_out,
    float* __restrict__ x_out) {
  extern __shared__ char smem[];
  float* sc = (float*)(smem + SC_OFF);
  bf16_t* abf = (bf16_t*)(smem + ABF_OFF);
  float* part = (float*)(smem + PART_OFF);
  float* red = (float*)(smem + RED_OFF);
  bf16_t* vstage = (bf16_t*)(smem + VST_OFF);

  const int tid = threadIdx.x;
  const int wave = tid >> 5;
  const int lane = tid & 31;
  const int lr = lane & 15;
  const int lh = lane >> 4;

  const int bh = blockIdx.x >> 6;  // b*H + h
  const int qt = blockIdx.x & 63;
  const int b = bh >> 3;
  const int h = bh & 7;
  const int q0 = qt << 4;

  const bf16_t* Qb = Q + (size_t)bh * Sn * Dp;
  const bf16_t* Kb = Kmat + (size_t)bh * Sn * Dp;
  const bf16_t* Vb = V + (size_t)bh * Sn * Dp;

#ifdef USE_TDM
  // ---- issue TDM: stage the whole V head [1024 x 64] bf16 into LDS, ----
  // ---- overlapped with score computation + softmax (phases 1-2).     ----
  {
    const unsigned long long ga = (unsigned long long)(const void*)Vb;
    const unsigned ldsa = (unsigned)(unsigned long long)(void*)vstage;
    u32x4 g0;
    g0[0] = 1u;                                  // count=1, user descriptor
    g0[1] = ldsa;                                // lds_addr
    g0[2] = (unsigned)(ga & 0xffffffffu);        // global_addr[95:64]
    g0[3] = (unsigned)((ga >> 32) & 0x1ffffffu)  // global_addr[120:96]
            | (2u << 30);                        // type = 2 (image)
    i32x8 g1;
    g1[0] = 0x00010000;                          // data_size=1 (2 bytes)
    g1[1] = (int)((Dp & 0xffff) << 16);          // tensor_dim0 lo16
    g1[2] = (int)(((Sn & 0xffff) << 16) | ((Dp >> 16) & 0xffff));
    g1[3] = (int)(((Dp & 0xffff) << 16) | ((Sn >> 16) & 0xffff));  // tile_dim0
    g1[4] = Sn;                                  // tile_dim1 = 1024 rows
    g1[5] = Dp;                                  // tensor_dim0_stride = 64
    g1[6] = 0;
    g1[7] = 0;
    i32x4 gz4;
    gz4[0] = gz4[1] = gz4[2] = gz4[3] = 0;
    i32x8 gz8;
#pragma unroll
    for (int i = 0; i < 8; ++i) gz8[i] = 0;
    if (wave == 0)
      __builtin_amdgcn_tensor_load_to_lds(g0, g1, gz4, gz4, gz8, 0);
  }
#endif

  // ---------------- Phase 1: scores = Q Kt / 8 + mask * -1e9 ---------------
  const bf16_t* qrow = Qb + (size_t)(q0 + lr) * Dp + lh * 16;
  v16bf a0 = frag_from_2x8(qrow, qrow + 8);         // depth 0..31
  v16bf a1 = frag_from_2x8(qrow + 32, qrow + 40);   // depth 32..63

  for (int kt = wave; kt < Sn / 16; kt += 8) {
    const int key0 = kt << 4;
    const bf16_t* krow = Kb + (size_t)(key0 + lr) * Dp + lh * 16;
    v16bf b0 = frag_from_2x8(krow, krow + 8);        // depth 0..31
    v16bf b1 = frag_from_2x8(krow + 32, krow + 40);  // depth 32..63

    v8f s = zero_v8f();
    s = wmma_bf16(a0, b0, s);
    s = wmma_bf16(a1, b1, s);

    const float mk = mask[(size_t)b * Sn + key0 + lr] * -1e9f;
#pragma unroll
    for (int r = 0; r < 8; ++r)
      sc[(lh * 8 + r) * Sn + key0 + lr] = s[r] * 0.125f + mk;
  }
  __syncthreads();

  // ---------------- Phase 2: softmax, * additional_weights -----------------
  const int row = tid >> 4;
  const int sub = tid & 15;

  float m = -3.402823466e+38f;
#pragma unroll
  for (int j = 0; j < 64; ++j) m = fmaxf(m, sc[row * Sn + sub + j * 16]);
  red[tid] = m;
  __syncthreads();
  if (sub == 0) {
    float mm = red[row * 16];
    for (int t = 1; t < 16; ++t) mm = fmaxf(mm, red[row * 16 + t]);
    red[row * 16] = mm;
  }
  __syncthreads();
  const float rowmax = red[row * 16];
  __syncthreads();

  float psum = 0.0f;
#pragma unroll
  for (int j = 0; j < 64; ++j) {
    const int idx = row * Sn + sub + j * 16;
    const float e = __expf(sc[idx] - rowmax);
    sc[idx] = e;
    psum += e;
  }
  red[tid] = psum;
  __syncthreads();
  if (sub == 0) {
    float ss = red[row * 16];
    for (int t = 1; t < 16; ++t) ss += red[row * 16 + t];
    red[row * 16] = ss;
  }
  __syncthreads();
  const float inv = 1.0f / red[row * 16];

  const float* aw = addw + ((size_t)b * Sn + q0 + row) * Sn;
  float* aout = attn_out + ((size_t)bh * Sn + q0 + row) * Sn;
#pragma unroll
  for (int j = 0; j < 64; ++j) {
    const int c = sub + j * 16;
    const float val = sc[row * Sn + c] * inv * aw[c];
    aout[c] = val;
    abf[row * Sn + c] = (bf16_t)val;
  }

#ifdef USE_TDM
  __builtin_amdgcn_s_wait_tensorcnt(0);  // wave0's TDM copy complete
#endif
  __syncthreads();  // abf ready + V staged for everyone

  // ---------------- Phase 3: x(16x64) = attn(16x1024) @ V(1024x64) ---------
  const int ntile = wave & 3;
  const int half = wave >> 2;

  v8f acc = zero_v8f();
  for (int k0 = half * 512; k0 < half * 512 + 512; k0 += 32) {
    const bf16_t* arow = abf + (size_t)lr * Sn + k0 + lh * 16;
    v16bf a = frag_from_2x8(arow, arow + 8);

    v16bf bv;
#ifdef USE_TDM
    const bf16_t* vp = vstage + (size_t)(k0 + lh * 16) * Dp + ntile * 16 + lr;
#else
    const bf16_t* vp = Vb + (size_t)(k0 + lh * 16) * Dp + ntile * 16 + lr;
#endif
#pragma unroll
    for (int i = 0; i < 16; ++i) bv[i] = vp[(size_t)i * Dp];

    acc = wmma_bf16(a, bv, acc);
  }

  float* p = part + wave * 256;
#pragma unroll
  for (int r = 0; r < 8; ++r) p[(lh * 8 + r) * 16 + lr] = acc[r];
  __syncthreads();

  if (half == 0) {
    const float* p0 = part + ntile * 256;
    const float* p1 = part + (ntile + 4) * 256;
#pragma unroll
    for (int r = 0; r < 8; ++r) {
      const int rr = lh * 8 + r;
      const float val = p0[rr * 16 + lr] + p1[rr * 16 + lr];
      x_out[((size_t)b * Sn + q0 + rr) * Dm + h * Dp + ntile * 16 + lr] = val;
    }
  }
}

// ---------------------------------------------------------------------------
extern "C" void kernel_launch(void* const* d_in, const int* in_sizes, int n_in,
                              void* d_out, int out_size, void* d_ws,
                              size_t ws_size, hipStream_t stream) {
  const float* q_in = (const float*)d_in[0];
  const float* k_in = (const float*)d_in[1];
  const float* v_in = (const float*)d_in[2];
  const float* mask = (const float*)d_in[3];
  const float* addw = (const float*)d_in[4];
  const float* wq = (const float*)d_in[5];
  const float* bq = (const float*)d_in[6];
  const float* wk = (const float*)d_in[7];
  const float* bk = (const float*)d_in[8];
  const float* wv = (const float*)d_in[9];
  const float* bv = (const float*)d_in[10];
  const float* wo = (const float*)d_in[11];
  const float* bo = (const float*)d_in[12];

  float* out = (float*)d_out;                // [B,S,512]
  float* attn = out + (size_t)Bn * Sn * Dm;  // [B,H,S,S]

  char* ws = (char*)d_ws;
  const size_t qkvBytes = (size_t)Bn * Hn * Sn * Dp * sizeof(bf16_t);  // 8 MB
  bf16_t* Qbf = (bf16_t*)ws;
  bf16_t* Kbf = (bf16_t*)(ws + qkvBytes);
  bf16_t* Vbf = (bf16_t*)(ws + 2 * qkvBytes);
  float* x = (float*)(ws + 3 * qkvBytes);

  const int M = Bn * Sn;  // 8192
  dim3 gemmGrid(M / GM, Dm / GN);  // 64 x 8

  mha_gemm_bias_kernel<<<gemmGrid, 256, 0, stream>>>(
      q_in, wq, bq, nullptr, Qbf, M, Dm, Dm, 1);
  mha_gemm_bias_kernel<<<gemmGrid, 256, 0, stream>>>(
      k_in, wk, bk, nullptr, Kbf, M, Dm, Dm, 1);
  mha_gemm_bias_kernel<<<gemmGrid, 256, 0, stream>>>(
      v_in, wv, bv, nullptr, Vbf, M, Dm, Dm, 1);

  const int attnBlocks = Bn * Hn * (Sn / 16);  // 4096
  mha_attention_kernel<<<attnBlocks, 256, SMEM_TOTAL, stream>>>(
      Qbf, Kbf, Vbf, mask, addw, attn, x);

  mha_gemm_bias_kernel<<<gemmGrid, 256, 0, stream>>>(
      x, wo, bo, out, nullptr, M, Dm, Dm, 0);
}
